// MambaLayer_8143257994026
// MI455X (gfx1250) — compile-verified, matrix operand order checked by reading, not run on hardware
//
#include <hip/hip_runtime.h>

#define D_MODEL 512
#define D_STATE 16
#define D_CONV  4
#define D_INNER 1024
#define DT_RANK 32
#define BATCH   16
#define SEQ     2048
#define NROW    (BATCH * SEQ)     // 32768 rows (b,l) flattened

typedef __bf16 bf16_t;
typedef __attribute__((ext_vector_type(16))) __bf16 v16bf;
typedef __attribute__((ext_vector_type(8)))  __bf16 v8bf;
typedef __attribute__((ext_vector_type(8)))  float  v8f;
typedef __attribute__((ext_vector_type(4)))  int    v4i;

// ---------------------------------------------------------------------------
// CDNA5 data-mover paths (guarded so the file compiles on any toolchain)
// ---------------------------------------------------------------------------
#if __has_builtin(__builtin_amdgcn_tensor_load_to_lds)
#define HAVE_TDM 1
typedef unsigned int u32;
typedef __attribute__((ext_vector_type(4))) unsigned int u32x4;
typedef __attribute__((ext_vector_type(4))) int i32x4;
typedef __attribute__((ext_vector_type(8))) int i32x8;

// 2D bf16 tile DMA: rows x 32 elements, row stride in elements, dest LDS
// contiguous row-major [rows][32]. D# layout per CDNA5 ISA ch.8.
__device__ __forceinline__ void tdm_load_tile_bf16(
    const bf16_t* gsrc, unsigned lds_byte, int rows,
    unsigned long long row_stride_elems, unsigned tensor_d0,
    unsigned tensor_d1) {
    unsigned long long ga = (unsigned long long)(__SIZE_TYPE__)gsrc;
    u32x4 g0 = {1u,                         // count=1, user descriptor
                lds_byte,                   // lds_addr
                (u32)ga,                    // global_addr lo
                (u32)(((ga >> 32) & 0x01FFFFFFull) | (2u << 30))};  // type=2
    i32x8 g1;
    g1[0] = 1 << 16;                                     // data_size=2B, mask=0
    g1[1] = (int)((tensor_d0 & 0xFFFFu) << 16);          // tensor_dim0 lo16
    g1[2] = (int)((tensor_d0 >> 16) | ((tensor_d1 & 0xFFFFu) << 16));
    g1[3] = (int)((tensor_d1 >> 16) | (32u << 16));      // tile_dim0 = 32
    g1[4] = rows;                                        // tile_dim1, tile_dim2=0
    g1[5] = (int)(u32)row_stride_elems;                  // dim0 stride lo32
    g1[6] = (int)((u32)(row_stride_elems >> 32) & 0xFFFFu);
    g1[7] = 0;
    i32x4 z4 = (i32x4)0;
#if __clang_major__ >= 23
    i32x8 z8 = (i32x8)0;
    __builtin_amdgcn_tensor_load_to_lds(g0, g1, z4, z4, z8, 0);
#else
    __builtin_amdgcn_tensor_load_to_lds(g0, g1, z4, z4, 0);
#endif
}
#endif  // HAVE_TDM

#if __has_builtin(__builtin_amdgcn_global_load_async_to_lds_b128)
#define HAVE_ASYNC 1
__device__ __forceinline__ void async_copy_b128(const void* g, void* l) {
    __builtin_amdgcn_global_load_async_to_lds_b128((v4i*)g, (v4i*)l, 0, 0);
}
__device__ __forceinline__ void async_wait0() {
#if __has_builtin(__builtin_amdgcn_s_wait_asynccnt)
    __builtin_amdgcn_s_wait_asynccnt(0);
#else
    asm volatile("s_wait_asynccnt 0x0" ::: "memory");
#endif
}
#endif  // HAVE_ASYNC

__device__ __forceinline__ unsigned lds_offset_of(const void* p) {
    // Generic LDS addresses carry the workgroup-relative byte offset in the
    // low 32 bits (aperture lives in addr[63:32], ISA 10.2).
    return (unsigned)(__SIZE_TYPE__)p;
}

// ---------------------------------------------------------------------------
// f32 -> bf16 conversion (grid-stride)
// ---------------------------------------------------------------------------
__global__ void cvt_f32_bf16_kernel(const float* __restrict__ in,
                                    bf16_t* __restrict__ out, int n) {
    for (int i = blockIdx.x * blockDim.x + threadIdx.x; i < n;
         i += gridDim.x * blockDim.x) {
        out[i] = (bf16_t)in[i];
    }
}

// ---------------------------------------------------------------------------
// Tiled bf16 GEMM with WMMA:  C[M,N] = A[M,K] * B[N,K]^T  (f32 accumulate)
// Block: 128 x NBLK, 256 threads (8 waves as 4x2), K-step 32.
// Tile staging via TDM (double-buffered) when available.
// ---------------------------------------------------------------------------
template <int NBLK>
__global__ __launch_bounds__(256) void gemm_bf16_kernel(
    const bf16_t* __restrict__ A,   // [M,K] row-major
    const bf16_t* __restrict__ B,   // [N,K] row-major (weight)
    float* __restrict__ C,          // [M,N] row-major
    int M, int N, int K) {
    constexpr int MBLK = 128;
    constexpr int WTM  = 2;                 // 16x16 tiles per wave in M
    constexpr int WTN  = NBLK / 32;         // 16x16 tiles per wave in N

    __shared__ __align__(16) bf16_t As[2][MBLK * 32];
    __shared__ __align__(16) bf16_t Bs[2][NBLK * 32];

    const int tid   = threadIdx.x;
    const int lane  = tid & 31;
    const int w     = tid >> 5;             // 0..7
    const int wm    = w & 3;                // wave row  (4)
    const int wn    = w >> 2;               // wave col  (2)
    const int mlane = lane & 15;
    const int hlf   = lane >> 4;

    const int m0 = blockIdx.y * MBLK;
    const int n0 = blockIdx.x * NBLK;

    v8f acc[WTM][WTN];
    for (int i = 0; i < WTM; ++i)
        for (int j = 0; j < WTN; ++j)
            for (int e = 0; e < 8; ++e) acc[i][j][e] = 0.0f;

    const int nk = K / 32;

#ifdef HAVE_TDM
    const bool issuer = (tid < 32);   // wave 0 drives the DMA engine
    if (issuer) {
        tdm_load_tile_bf16(&A[(size_t)m0 * K], lds_offset_of(&As[0][0]), MBLK,
                           (unsigned long long)K, (unsigned)K, (unsigned)M);
        tdm_load_tile_bf16(&B[(size_t)n0 * K], lds_offset_of(&Bs[0][0]), NBLK,
                           (unsigned long long)K, (unsigned)K, (unsigned)N);
    }
#endif

    for (int kt = 0; kt < nk; ++kt) {
        const int cur = kt & 1;

#ifdef HAVE_TDM
        if (issuer) {
            if (kt + 1 < nk) {
                const int k1 = (kt + 1) * 32;
                tdm_load_tile_bf16(&A[(size_t)m0 * K + k1],
                                   lds_offset_of(&As[1 - cur][0]), MBLK,
                                   (unsigned long long)K, (unsigned)K,
                                   (unsigned)M);
                tdm_load_tile_bf16(&B[(size_t)n0 * K + k1],
                                   lds_offset_of(&Bs[1 - cur][0]), NBLK,
                                   (unsigned long long)K, (unsigned)K,
                                   (unsigned)N);
                __builtin_amdgcn_s_wait_tensorcnt(2);  // current buffer ready
            } else {
                __builtin_amdgcn_s_wait_tensorcnt(0);
            }
        }
        __syncthreads();
#else
        // Fallback: register staging (NBLK=128 loads b1 after the other stores).
        const int k0 = kt * 32;
        __syncthreads();
        {
            const int c0 = tid, c1 = tid + 256;  // MBLK*4 = 512 chunks
            v8bf a0 = *(const v8bf*)&A[(size_t)(m0 + (c0 >> 2)) * K + k0 +
                                       (c0 & 3) * 8];
            v8bf a1 = *(const v8bf*)&A[(size_t)(m0 + (c1 >> 2)) * K + k0 +
                                       (c1 & 3) * 8];
            v8bf b0 = *(const v8bf*)&B[(size_t)(n0 + (c0 >> 2)) * K + k0 +
                                       (c0 & 3) * 8];
            *(v8bf*)&As[cur][(c0 >> 2) * 32 + (c0 & 3) * 8] = a0;
            *(v8bf*)&As[cur][(c1 >> 2) * 32 + (c1 & 3) * 8] = a1;
            *(v8bf*)&Bs[cur][(c0 >> 2) * 32 + (c0 & 3) * 8] = b0;
            if (NBLK == 128) {
                v8bf b1 = *(const v8bf*)&B[(size_t)(n0 + (c1 >> 2)) * K + k0 +
                                           (c1 & 3) * 8];
                *(v8bf*)&Bs[cur][(c1 >> 2) * 32 + (c1 & 3) * 8] = b1;
            }
        }
        __syncthreads();
#endif

        // Per-lane fragment gather (ISA 16-bit A layout: lanes 0-15 hold
        // K{0..7,16..23}, lanes 16-31 hold K{8..15,24..31} for row M=lane%16)
        v16bf afr[WTM], bfr[WTN];
        for (int tm = 0; tm < WTM; ++tm) {
            const bf16_t* p =
                &As[cur][(wm * 32 + tm * 16 + mlane) * 32 + hlf * 8];
            v8bf lo = *(const v8bf*)p;
            v8bf hi = *(const v8bf*)(p + 16);
            afr[tm] = __builtin_shufflevector(lo, hi, 0, 1, 2, 3, 4, 5, 6, 7,
                                              8, 9, 10, 11, 12, 13, 14, 15);
        }
        for (int tn = 0; tn < WTN; ++tn) {
            const bf16_t* p =
                &Bs[cur][(wn * (NBLK / 2) + tn * 16 + mlane) * 32 + hlf * 8];
            v8bf lo = *(const v8bf*)p;
            v8bf hi = *(const v8bf*)(p + 16);
            bfr[tn] = __builtin_shufflevector(lo, hi, 0, 1, 2, 3, 4, 5, 6, 7,
                                              8, 9, 10, 11, 12, 13, 14, 15);
        }
        for (int tm = 0; tm < WTM; ++tm)
            for (int tn = 0; tn < WTN; ++tn)
                acc[tm][tn] = __builtin_amdgcn_wmma_f32_16x16x32_bf16(
                    false, afr[tm], false, bfr[tn], (short)0, acc[tm][tn],
                    false, false);

#ifdef HAVE_TDM
        __syncthreads();  // all waves done reading before DMA overwrites
#endif
    }

    // Epilogue: C/D layout: VGPR r, lane l -> M = r + 8*(l>>4), N = l&15
    for (int tm = 0; tm < WTM; ++tm)
        for (int tn = 0; tn < WTN; ++tn) {
            int gm = m0 + wm * 32 + tm * 16 + hlf * 8;
            int gn = n0 + wn * (NBLK / 2) + tn * 16 + mlane;
#pragma unroll
            for (int r = 0; r < 8; ++r)
                C[(size_t)(gm + r) * N + gn] = acc[tm][tn][r];
        }
}

// ---------------------------------------------------------------------------
// Depthwise causal conv (k=4) + SiLU, bf16 output for x_proj GEMM A operand.
// xc is columns [0,1024) of xz [NROW, 2048].
// ---------------------------------------------------------------------------
__global__ __launch_bounds__(256) void conv_silu_kernel(
    const float* __restrict__ xz, const float* __restrict__ conv_w,
    const float* __restrict__ conv_b, bf16_t* __restrict__ xa_bf) {
    int idx = blockIdx.x * blockDim.x + threadIdx.x;  // over NROW * D_INNER
    int d  = idx & (D_INNER - 1);
    int bl = idx >> 10;
    int l  = bl & (SEQ - 1);
    int b  = bl >> 11;
    float acc = conv_b[d];
#pragma unroll
    for (int k = 0; k < D_CONV; ++k) {
        int ls = l - (D_CONV - 1) + k;
        if (ls >= 0)
            acc += xz[(size_t)(b * SEQ + ls) * (2 * D_INNER) + d] *
                   conv_w[d * D_CONV + k];
    }
    float s = acc / (1.0f + __expf(-acc));  // SiLU
    xa_bf[(size_t)bl * D_INNER + d] = (bf16_t)s;
}

__device__ __forceinline__ float softplus_f(float x) {
    return (x > 20.0f) ? x : log1pf(__expf(x));
}

// ---------------------------------------------------------------------------
// Selective scan, fused with dt_proj(+softplus) and silu(z) output gate.
// One thread per (b, d) channel: 16-state recurrence over L in registers.
// dbl[b, t, 0:64] chunks staged in LDS via async global->LDS copies.
// ---------------------------------------------------------------------------
__global__ __launch_bounds__(256) void scan_kernel(
    const float* __restrict__ xz,       // [NROW, 2048]; z = cols [1024,2048)
    const bf16_t* __restrict__ xa_bf,   // [NROW, 1024]
    const float* __restrict__ dbl,      // [NROW, 64]
    const float* __restrict__ w_dt,     // [1024, 32]
    const float* __restrict__ b_dt,     // [1024]
    const float* __restrict__ A_log,    // [1024, 16]
    const float* __restrict__ Dp,       // [1024]
    bf16_t* __restrict__ y_bf) {        // [NROW, 1024]
    const int b   = blockIdx.x >> 2;
    const int dc  = blockIdx.x & 3;
    const int tid = threadIdx.x;
    const int d   = dc * 256 + tid;

    float dtw[DT_RANK];
#pragma unroll
    for (int r = 0; r < DT_RANK; ++r) dtw[r] = w_dt[d * DT_RANK + r];
    const float bias = b_dt[d];
    float Ar[D_STATE];
#pragma unroll
    for (int n = 0; n < D_STATE; ++n) Ar[n] = -__expf(A_log[d * D_STATE + n]);
    const float Dv = Dp[d];
    float h[D_STATE];
#pragma unroll
    for (int n = 0; n < D_STATE; ++n) h[n] = 0.0f;

    __shared__ __align__(16) float sdbl[32][64];  // 8 KB chunk of dbl

    for (int t0 = 0; t0 < SEQ; t0 += 32) {
        __syncthreads();
        {
            const int i0 = tid, i1 = tid + 256;  // 512 float4 chunks
            const int r0 = i0 >> 4, c0 = i0 & 15;
            const int r1 = i1 >> 4, c1 = i1 & 15;
            const float* g0 = &dbl[(size_t)(b * SEQ + t0 + r0) * 64 + c0 * 4];
            const float* g1 = &dbl[(size_t)(b * SEQ + t0 + r1) * 64 + c1 * 4];
#ifdef HAVE_ASYNC
            async_copy_b128(g0, &sdbl[r0][c0 * 4]);
            async_copy_b128(g1, &sdbl[r1][c1 * 4]);
            async_wait0();
#else
            *(float4*)&sdbl[r0][c0 * 4] = *(const float4*)g0;
            *(float4*)&sdbl[r1][c1 * 4] = *(const float4*)g1;
#endif
        }
        __syncthreads();

        for (int tt = 0; tt < 32; ++tt) {
            const int row = b * SEQ + t0 + tt;
            float dtv = bias;
#pragma unroll
            for (int r = 0; r < DT_RANK; ++r) dtv += sdbl[tt][r] * dtw[r];
            dtv = softplus_f(dtv);

            const float u  = (float)xa_bf[(size_t)row * D_INNER + d];
            const float du = dtv * u;
            float y = 0.0f;
#pragma unroll
            for (int n = 0; n < D_STATE; ++n) {
                float dA = __expf(dtv * Ar[n]);
                h[n] = dA * h[n] + du * sdbl[tt][DT_RANK + n];
                y += h[n] * sdbl[tt][DT_RANK + D_STATE + n];
            }
            y += u * Dv;

            const float z  = xz[(size_t)row * (2 * D_INNER) + D_INNER + d];
            const float yz = y * (z / (1.0f + __expf(-z)));
            y_bf[(size_t)row * D_INNER + d] = (bf16_t)yz;
        }
    }
}

// ---------------------------------------------------------------------------
// Residual + LayerNorm over d_model=512; one block (256 threads) per row.
// ---------------------------------------------------------------------------
__global__ __launch_bounds__(256) void ln_kernel(
    const float* __restrict__ hp, const float* __restrict__ xin,
    const float* __restrict__ g, const float* __restrict__ be,
    float* __restrict__ out) {
    const int row = blockIdx.x;
    const int tid = threadIdx.x;
    __shared__ float red[256];
    const size_t base = (size_t)row * D_MODEL;

    float v0 = hp[base + tid] + xin[base + tid];
    float v1 = hp[base + 256 + tid] + xin[base + 256 + tid];

    red[tid] = v0 + v1;
    __syncthreads();
    for (int off = 128; off > 0; off >>= 1) {
        if (tid < off) red[tid] += red[tid + off];
        __syncthreads();
    }
    const float mu = red[0] * (1.0f / D_MODEL);
    __syncthreads();

    const float d0 = v0 - mu, d1 = v1 - mu;
    red[tid] = d0 * d0 + d1 * d1;
    __syncthreads();
    for (int off = 128; off > 0; off >>= 1) {
        if (tid < off) red[tid] += red[tid + off];
        __syncthreads();
    }
    const float rstd = rsqrtf(red[0] * (1.0f / D_MODEL) + 1e-12f);

    out[base + tid]       = d0 * rstd * g[tid] + be[tid];
    out[base + 256 + tid] = d1 * rstd * g[tid + 256] + be[tid + 256];
}

// ---------------------------------------------------------------------------
extern "C" void kernel_launch(void* const* d_in, const int* in_sizes, int n_in,
                              void* d_out, int out_size, void* d_ws,
                              size_t ws_size, hipStream_t stream) {
    const float* x      = (const float*)d_in[0];   // [16,2048,512]
    const float* w_in   = (const float*)d_in[1];   // [2048,512]
    const float* conv_w = (const float*)d_in[2];   // [1024,4]
    const float* conv_b = (const float*)d_in[3];   // [1024]
    const float* w_xp   = (const float*)d_in[4];   // [64,1024]
    const float* w_dt   = (const float*)d_in[5];   // [1024,32]
    const float* b_dt   = (const float*)d_in[6];   // [1024]
    const float* A_log  = (const float*)d_in[7];   // [1024,16]
    const float* Dp     = (const float*)d_in[8];   // [1024]
    const float* w_out  = (const float*)d_in[9];   // [512,1024]
    const float* gamma  = (const float*)d_in[10];  // [512]
    const float* beta   = (const float*)d_in[11];  // [512]
    float* out = (float*)d_out;

    char* wsp = (char*)d_ws;
    auto alloc = [&](size_t bytes) -> char* {
        char* p = wsp;
        wsp += (bytes + 255) & ~(size_t)255;
        return p;
    };

    bf16_t* x_bf    = (bf16_t*)alloc((size_t)NROW * D_MODEL * 2);
    bf16_t* win_bf  = (bf16_t*)alloc((size_t)2 * D_INNER * D_MODEL * 2);
    bf16_t* wxp_bf  = (bf16_t*)alloc((size_t)64 * D_INNER * 2);
    bf16_t* wout_bf = (bf16_t*)alloc((size_t)D_MODEL * D_INNER * 2);
    float*  xz      = (float*)alloc((size_t)NROW * 2 * D_INNER * 4);
    bf16_t* xa_bf   = (bf16_t*)alloc((size_t)NROW * D_INNER * 2);
    float*  dbl     = (float*)alloc((size_t)NROW * 64 * 4);
    bf16_t* y_bf    = (bf16_t*)alloc((size_t)NROW * D_INNER * 2);
    float*  hpre    = (float*)alloc((size_t)NROW * D_MODEL * 4);

    // 1) bf16 conversions
    cvt_f32_bf16_kernel<<<4096, 256, 0, stream>>>(x, x_bf, NROW * D_MODEL);
    cvt_f32_bf16_kernel<<<1024, 256, 0, stream>>>(w_in, win_bf,
                                                  2 * D_INNER * D_MODEL);
    cvt_f32_bf16_kernel<<<64, 256, 0, stream>>>(w_xp, wxp_bf, 64 * D_INNER);
    cvt_f32_bf16_kernel<<<512, 256, 0, stream>>>(w_out, wout_bf,
                                                 D_MODEL * D_INNER);

    // 2) in_proj: xz = x @ w_in^T  (M=32768, N=2048, K=512)
    gemm_bf16_kernel<128><<<dim3(2 * D_INNER / 128, NROW / 128), 256, 0,
                            stream>>>(x_bf, win_bf, xz, NROW, 2 * D_INNER,
                                      D_MODEL);

    // 3) conv + silu -> xa (bf16)
    conv_silu_kernel<<<(NROW * D_INNER) / 256, 256, 0, stream>>>(
        xz, conv_w, conv_b, xa_bf);

    // 4) x_proj: dbl = xa @ w_xp^T  (M=32768, N=64, K=1024)
    gemm_bf16_kernel<64><<<dim3(1, NROW / 128), 256, 0, stream>>>(
        xa_bf, wxp_bf, dbl, NROW, 64, D_INNER);

    // 5) fused dt_proj + selective scan + silu(z) gate -> y (bf16)
    scan_kernel<<<BATCH * 4, 256, 0, stream>>>(xz, xa_bf, dbl, w_dt, b_dt,
                                               A_log, Dp, y_bf);

    // 6) out_proj: hpre = y @ w_out^T  (M=32768, N=512, K=1024)
    gemm_bf16_kernel<128><<<dim3(D_MODEL / 128, NROW / 128), 256, 0, stream>>>(
        y_bf, wout_bf, hpre, NROW, D_MODEL, D_INNER);

    // 7) residual + layernorm -> out
    ln_kernel<<<NROW, 256, 0, stream>>>(hpre, x, gamma, beta, out);
}
